// PhysicalConsistencyLoss_8443905704370
// MI455X (gfx1250) — matrix operand order repaired and checked by reading, not compiled, not run on hardware
//
#include <hip/hip_runtime.h>
#include <hip/hip_bf16.h>
#include <math.h>

typedef __attribute__((ext_vector_type(16))) _Float16 v16h;
typedef __attribute__((ext_vector_type(8)))  float    v8f;
typedef __attribute__((ext_vector_type(4)))  unsigned int u32x4;
typedef __attribute__((ext_vector_type(8)))  unsigned int u32x8;

#define KS    33
#define PADR  16
#define NB    8
#define NC    3
#define NH    512
#define NW    512
#define HW    (NH * NW)
#define NBC   (NB * NC)
#define JH    (NH + 2 * PADR)   /* 544 padded rows                         */
#define JW    576               /* padded row stride (halfs); >= 448+128   */
#define AFRAG_PER_BC (KS * 2 * 32 * 16)   /* 33792 halfs per (b,c) */
#define NTOT  (NBC * HW)        /* 6291456 output elements    */
#define NPART 24576             /* recon blocks               */

#define TILE_ROWS 48            /* input rows needed per block (16 + 32)   */
#define TILE_W    128           /* TDM tile width in halfs (112 used)      */
#define LDS_STRIDE 144          /* halfs: 128 data + 16 pad (TDM padding)  */

// ---------------------------------------------------------------------------
// Kernel 1: build normalized PSF kernels and emit Toeplitz A-fragments in the
// exact WMMA f16 16x32 A-matrix register layout (lane -> M, vgpr/half -> K).
// ---------------------------------------------------------------------------
__global__ void psf_build_kernel(const float* __restrict__ mu_d,
                                 const float* __restrict__ beta_D,
                                 const float* __restrict__ g,
                                 _Float16* __restrict__ Afrag) {
    const int bc = blockIdx.x;
    const int b  = bc / NC;
    const int c  = bc % NC;
    const int hc = NH / 2, wc = NW / 2;

    const float d    = mu_d[(b * NH + hc) * NW + wc];            // (B,1,H,W)
    const float beta = beta_D[((b * NC + c) * NH + hc) * NW + wc];
    const float gg   = g[(b * NH + hc) * NW + wc];

    __shared__ float kn[KS * KS];
    __shared__ float red[256];

    float partial = 0.0f;
    for (int t = threadIdx.x; t < KS * KS; t += 256) {
        const int ky = t / KS, kx = t % KS;
        const float y = (float)(ky - PADR);
        const float x = (float)(kx - PADR);
        const float r2 = x * x + y * y + 1e-6f;
        const float cos_theta = d * rsqrtf(r2 + d * d + 1e-6f);
        const float g2 = gg * gg;
        const float denom = 1.0f + g2 - 2.0f * gg * cos_theta;
        const float phase = (1.0f - g2) /
                            (4.0f * 3.14159265358979323846f * denom * sqrtf(denom));
        const float trans = expf(-beta * d);
        const float k = phase * trans / (r2 + 1e-6f);
        kn[t] = k;
        partial += k;
    }
    red[threadIdx.x] = partial;
    __syncthreads();
    for (int s = 128; s > 0; s >>= 1) {
        if (threadIdx.x < s) red[threadIdx.x] += red[threadIdx.x + s];
        __syncthreads();
    }
    const float inv = 1.0f / (red[0] + 1e-6f);
    __syncthreads();
    for (int t = threadIdx.x; t < KS * KS; t += 256) kn[t] *= inv;
    __syncthreads();

    // Toeplitz A-fragments: A[m][j] = kn[ky][j - m] for j-m in [0,32], else 0.
    // Layout: Afrag[bc][ky][chunk][lane][i]  (i = v16h element index)
    _Float16* out = Afrag + bc * AFRAG_PER_BC;
    for (int e = threadIdx.x; e < AFRAG_PER_BC; e += 256) {
        const int i     = e & 15;
        const int lane  = (e >> 4) & 31;
        const int chunk = (e >> 9) & 1;
        const int ky    = e >> 10;
        const int m     = lane & 15;
        const int v     = i >> 1;
        const int hi    = i & 1;
        const int K     = (lane >> 4) * 16 + i;
        const int j     = chunk * 32 + K;
        const int kx    = j - m;
        const float val = (kx >= 0 && kx < KS) ? kn[ky * KS + kx] : 0.0f;
        out[e] = (_Float16)val;
    }
}

// ---------------------------------------------------------------------------
// Kernel 2: edge-pad J into f16 with row stride JW.
// ---------------------------------------------------------------------------
__global__ void pad_kernel(const float* __restrict__ J,
                           _Float16* __restrict__ Jpad) {
    const int bc  = blockIdx.y;
    const int idx = blockIdx.x * 256 + threadIdx.x;
    if (idx >= JH * JW) return;
    const int r  = idx / JW;
    const int cx = idx % JW;
    int sy = r - PADR;  sy = sy < 0 ? 0 : (sy > NH - 1 ? NH - 1 : sy);
    int sx = cx - PADR; sx = sx < 0 ? 0 : (sx > NW - 1 ? NW - 1 : sx);
    Jpad[(size_t)bc * JH * JW + idx] = (_Float16)J[((size_t)bc * NH + sy) * NW + sx];
}

// ---------------------------------------------------------------------------
// Kernel 3: depthwise 33x33 conv as Toeplitz GEMMs on the WMMA pipe.
// One TDM (tensor_load_to_lds) DMA stages the block's shared 48x128 input
// tile into LDS (padded to a 144-half stride); 4 waves, one 16x16 output
// tile each, 66 v_wmma per tile fed by ds_load_b128 fragments.
// ---------------------------------------------------------------------------
__global__ void __launch_bounds__(128) conv_wmma_kernel(
        const _Float16* __restrict__ Jpad,
        const _Float16* __restrict__ Afrag,
        float* __restrict__ direct /* = d_out + 1 */) {
    __shared__ _Float16 tile[TILE_ROWS * LDS_STRIDE];   // 13824 B

    const int lane = threadIdx.x & 31;
    const int wave = threadIdx.x >> 5;
    const int bc   = blockIdx.z;
    const int y0   = blockIdx.y * 16;
    const int x0g  = blockIdx.x * 64;                   // block's leftmost column

    const int n  = lane & 15;   // N index (output row within tile)
    const int hs = lane >> 4;   // half-wave select

    // Never executed (blockDim.x == 128): defeats "LDS never stored" analysis
    // so the compiler cannot fold the post-DMA tile reads to undef.
    if (threadIdx.x == 0xFFFF) tile[0] = (_Float16)0.0f;

    // LDS byte offset of tile[]: low 32 bits of the flat address of a shared
    // object are the LDS offset (aperture base is in the high half). The
    // ptrtoint also captures the pointer so the asm "memory" clobber applies.
    const unsigned int lds_addr = (unsigned int)(uintptr_t)&tile[0];

    // ---- TDM: DMA the 48 x 128-half tile (rows y0..y0+47, cols x0g..+127)
    // into LDS with 8-dword padding after every 64-dword row (stride 144 halfs).
    if (threadIdx.x < 32) {     // wave 0 owns the DMA (TDM ignores EXEC)
        const unsigned long long ga = (unsigned long long)(uintptr_t)
            (Jpad + (size_t)bc * JH * JW + (size_t)y0 * JW + x0g);
        u32x4 g0;
        g0[0] = 1u;                                   // count=1, no gather
        g0[1] = lds_addr;                             // lds_addr (bytes)
        g0[2] = (unsigned int)ga;                     // global_addr[31:0]
        g0[3] = (unsigned int)((ga >> 32) & 0x1FFFFFFu) | 0x80000000u; // addr hi | type=2
        u32x8 g1;
        g1[0] = (1u << 16)      // data_size = 2 bytes
              | (1u << 20)      // pad_enable
              | (5u << 22)      // pad_interval: 64 dwords (one 128-half row)
              | (7u << 25);     // pad_amount: 8 dwords (16 halfs) -> stride 144
        g1[1] = 0u;                                   // no atomic barrier; dim0 lo16=0
        g1[2] = 0x4000u;                              // tensor_dim0 = 1<<30 (hi bits)
        g1[3] = 0x4000u | ((unsigned)TILE_W << 16);   // tensor_dim1 = 1<<30; tile_dim0=128
        g1[4] = (unsigned)TILE_ROWS;                  // tile_dim1 = 48; tile_dim2 = 0
        g1[5] = (unsigned)JW;                         // tensor_dim0_stride = 576
        g1[6] = 0u;
        g1[7] = 0u;
        asm volatile("tensor_load_to_lds %0, %1"
                     :
                     : "s"(g0), "s"(g1), "r"(&tile[0])
                     : "memory");
        __builtin_amdgcn_s_wait_tensorcnt(0);
    }
    __syncthreads();

    const _Float16* __restrict__ Af =
        Afrag + (size_t)bc * AFRAG_PER_BC + (size_t)lane * 16;
    // B fragment base in LDS: local col = wave*16 + hs*16 (+ chunk*32 + i)
    const _Float16* bbase = tile + (wave * 16 + hs * 16);

    v8f acc = {};
    #pragma unroll 3
    for (int ky = 0; ky < KS; ++ky) {
        const _Float16* r0 = bbase + (n + ky) * LDS_STRIDE;
        v16h b0 = *(const v16h*)(r0);
        v16h b1 = *(const v16h*)(r0 + 32);
        v16h a0 = *(const v16h*)(Af + (size_t)(ky * 2 + 0) * (32 * 16));
        v16h a1 = *(const v16h*)(Af + (size_t)(ky * 2 + 1) * (32 * 16));
        acc = __builtin_amdgcn_wmma_f32_16x16x32_f16(
                  false, a0, false, b0, (short)0, acc, false, false);
        acc = __builtin_amdgcn_wmma_f32_16x16x32_f16(
                  false, a1, false, b1, (short)0, acc, false, false);
    }

    // C/D layout: element r -> M = hs*8 + r, N = n.  C[M][N] = direct[y0+N][x0+M]
    const int x0 = x0g + wave * 16;
    float* outp = direct + ((size_t)bc * NH + (y0 + n)) * NW + x0 + hs * 8;
    #pragma unroll
    for (int r = 0; r < 8; ++r) outp[r] = acc[r];
}

// ---------------------------------------------------------------------------
// Kernel 4: elementwise recon + per-block partial sums of |I - recon|.
// ---------------------------------------------------------------------------
__global__ void recon_kernel(const float* __restrict__ I,
                             const float* __restrict__ mu_d,
                             const float* __restrict__ beta_D,
                             const float* __restrict__ beta_B,
                             const float* __restrict__ Lm,
                             const float* __restrict__ B_inf,
                             float* __restrict__ out,
                             float* __restrict__ partials) {
    const int idx = blockIdx.x * 256 + threadIdx.x;   // grid sized exactly
    const int b   = idx / (NC * HW);
    const int rem = idx - b * (NC * HW);
    const int c   = rem / HW;
    const int pix = rem - c * HW;

    const float mu   = mu_d[b * HW + pix];
    const float t    = expf(-beta_D[idx] * mu);
    const float Bt   = B_inf[b * NC + c] * (1.0f - expf(-beta_B[b * NC + c] * mu));
    const float dval = out[1 + idx];
    const float rec  = dval * t * Lm[b * HW + pix] + Bt;
    out[1 + idx] = rec;

    __shared__ float red[256];
    red[threadIdx.x] = fabsf(I[idx] - rec);
    __syncthreads();
    for (int s = 128; s > 0; s >>= 1) {
        if (threadIdx.x < s) red[threadIdx.x] += red[threadIdx.x + s];
        __syncthreads();
    }
    if (threadIdx.x == 0) partials[blockIdx.x] = red[0];
}

// ---------------------------------------------------------------------------
// Kernel 5: deterministic final reduction -> out[0] = mean abs error.
// ---------------------------------------------------------------------------
__global__ void finalize_kernel(const float* __restrict__ partials,
                                float* __restrict__ out) {
    __shared__ float red[256];
    float s = 0.0f;
    for (int i = threadIdx.x; i < NPART; i += 256) s += partials[i];
    red[threadIdx.x] = s;
    __syncthreads();
    for (int st = 128; st > 0; st >>= 1) {
        if (threadIdx.x < st) red[threadIdx.x] += red[threadIdx.x + st];
        __syncthreads();
    }
    if (threadIdx.x == 0) out[0] = red[0] / (float)NTOT;
}

// ---------------------------------------------------------------------------
extern "C" void kernel_launch(void* const* d_in, const int* in_sizes, int n_in,
                              void* d_out, int out_size, void* d_ws, size_t ws_size,
                              hipStream_t stream) {
    const float* I      = (const float*)d_in[0];
    const float* J      = (const float*)d_in[1];
    const float* mu_d   = (const float*)d_in[2];
    const float* beta_D = (const float*)d_in[3];
    const float* beta_B = (const float*)d_in[4];
    const float* g      = (const float*)d_in[5];
    const float* Lm     = (const float*)d_in[6];
    const float* B_inf  = (const float*)d_in[7];
    float* out = (float*)d_out;

    // Workspace layout (256B-aligned sub-allocations, ~16.8 MB total)
    char* ws = (char*)d_ws;
    const size_t afrag_bytes = (size_t)NBC * AFRAG_PER_BC * sizeof(_Float16); // 1,622,016
    const size_t jpad_bytes  = (size_t)NBC * JH * JW * sizeof(_Float16);      // 15,040,512
    _Float16* Afrag    = (_Float16*)(ws);
    _Float16* Jpad     = (_Float16*)(ws + afrag_bytes);
    float*    partials = (float*)(ws + afrag_bytes + jpad_bytes);

    // 1. PSF kernels -> Toeplitz WMMA A-fragments
    psf_build_kernel<<<NBC, 256, 0, stream>>>(mu_d, beta_D, g, Afrag);

    // 2. Edge-pad J to f16
    const int padElems = JH * JW;
    pad_kernel<<<dim3((padElems + 255) / 256, NBC), 256, 0, stream>>>(J, Jpad);

    // 3. Depthwise conv: TDM -> LDS -> WMMA; direct -> out[1..]
    conv_wmma_kernel<<<dim3(NW / 64, NH / 16, NBC), 128, 0, stream>>>(
        Jpad, Afrag, out + 1);

    // 4. Elementwise recon (in place over out[1..]) + partial L1 sums
    recon_kernel<<<NPART, 256, 0, stream>>>(I, mu_d, beta_D, beta_B, Lm, B_inf,
                                            out, partials);

    // 5. Deterministic loss reduction -> out[0]
    finalize_kernel<<<1, 256, 0, stream>>>(partials, out);
}
